// CondRnnSampler_35270271434835
// MI455X (gfx1250) — matrix operand order repaired and checked by reading, not compiled, not run on hardware
//
#include <hip/hip_runtime.h>

// ---- problem constants (B,D,E,NC) = (4096, 64, 256, 256) ----
#define BB 4096
#define DD 64
#define EE 256
#define H2 512
#define NCC 256
#define G4 1024
#define BT 32               // batch rows per workgroup (two WMMA M-tiles)

typedef __attribute__((ext_vector_type(16))) __bf16        v16bf;
typedef __attribute__((ext_vector_type(8)))  float         v8f;
typedef __attribute__((ext_vector_type(4)))  unsigned int  u32x4;

union Frag { v16bf v; u32x4 q[2]; };

__device__ __forceinline__ unsigned short f2bf(float f) {
    unsigned int u = __float_as_uint(f);
    u += 0x7FFFu + ((u >> 16) & 1u);            // round-to-nearest-even
    return (unsigned short)(u >> 16);
}
__device__ __forceinline__ float bf2f(unsigned short h) {
    return __uint_as_float(((unsigned int)h) << 16);
}
__device__ __forceinline__ float sigf(float x) {
    return 1.0f / (1.0f + __expf(-x));
}
__device__ __forceinline__ float tanhfast(float x) {
    x = fminf(fmaxf(x, -15.0f), 15.0f);
    float e = __expf(2.0f * x);
    return (e - 1.0f) / (e + 1.0f);
}
__device__ __forceinline__ v8f wmma_bf16(v16bf a, v16bf b, v8f c) {
    return __builtin_amdgcn_wmma_f32_16x16x32_bf16(
        /*neg_a=*/false, a, /*neg_b=*/false, b,
        /*c_mod=*/(short)0, c, /*reuse_a=*/false, /*reuse_b=*/false);
}

// ---- bf16 weight workspace layout in d_ws (ushort elements) ----
#define OFF_WIH 0
#define OFF_WHH (OFF_WIH + G4*EE)     // 262144
#define OFF_W1  (OFF_WHH + G4*EE)     // 524288
#define OFF_W2  (OFF_W1  + H2*EE)     // 655360
#define OFF_TOK (OFF_W2  + NCC*H2)    // 786432
#define WS_TOT  (OFF_TOK + NCC*EE)    // 851968 ushorts = 1,703,936 B

__global__ __launch_bounds__(256)
void cvt_weights(const float* __restrict__ wih, const float* __restrict__ whh,
                 const float* __restrict__ w1,  const float* __restrict__ w2,
                 const float* __restrict__ tok, unsigned short* __restrict__ ws) {
    int i = blockIdx.x * 256 + threadIdx.x;
    if (i >= WS_TOT) return;
    float v;
    if      (i < OFF_WHH) v = wih[i - OFF_WIH];
    else if (i < OFF_W1 ) v = whh[i - OFF_WHH];
    else if (i < OFF_W2 ) v = w1 [i - OFF_W1 ];
    else if (i < OFF_TOK) v = w2 [i - OFF_W2 ];
    else                  v = tok[i - OFF_TOK];
    ws[i] = f2bf(v);
}

// 8 waves/WG must co-reside 2-per-SIMD32 on one WGP: force the register
// budget for >=2 waves/SIMD so the workgroup is guaranteed schedulable.
__global__ __launch_bounds__(256, 2)
void rnn_persistent(const unsigned short* __restrict__ wih_bf,
                    const unsigned short* __restrict__ whh_bf,
                    const unsigned short* __restrict__ w1_bf,
                    const unsigned short* __restrict__ w2_bf,
                    const unsigned short* __restrict__ tok_bf,
                    const float* __restrict__ b_ih, const float* __restrict__ b_hh,
                    const float* __restrict__ b1,  const float* __restrict__ b2,
                    const int* __restrict__ pos_list, const int* __restrict__ samples,
                    float* __restrict__ out) {
    // ---- LDS (≈56 KB) ----
    __shared__ __align__(16) unsigned short hbf[BT][EE];       // 16 KB hidden state (bf16)
    __shared__ __align__(16) unsigned char  buf[BT * H2 * 2];  // 32 KB time-shared buffer
    __shared__ float red[BT][8];                               //  1 KB softmax partials
    __shared__ float rowmax[BT], logp[BT];
    __shared__ int   s_lds[BT];
    __shared__ float bsum[G4], b1s[H2], b2s[NCC];              //  7 KB biases

    // buf time-sharing: hid(bf16 32KB) -> logits(f32 32KB) -> x(bf16 lo 16KB) + h_new staging (hi 16KB)
    unsigned short (*hidbf)[H2]   = (unsigned short(*)[H2])buf;
    float          (*logits)[NCC] = (float(*)[NCC])buf;
    unsigned short (*xbf)[EE]     = (unsigned short(*)[EE])buf;
    unsigned short (*hnew)[EE]    = (unsigned short(*)[EE])(buf + BT * EE * 2);

    const int tid  = threadIdx.x;
    const int lane = tid & 31;
    const int wv   = tid >> 5;        // wave id 0..7
    const int half = lane >> 4;       // 0/1 half-wave (WMMA layout)
    const int am   = lane & 15;       // A row / B col within 16-tile
    const int r0   = blockIdx.x * BT; // first batch row of this WG

    // cell state lives in registers: c element (m,col) is owned by exactly one
    // (lane, accumulator-vgpr) => c_reg[t-pass][mm][v]
    float c_reg[2][2][8] = {};

    // ---- preload biases, zero hidden state ----
    for (int i = tid; i < G4;  i += 256) bsum[i] = b_ih[i] + b_hh[i];
    for (int i = tid; i < H2;  i += 256) b1s[i]  = b1[i];
    for (int i = tid; i < NCC; i += 256) b2s[i]  = b2[i];
    for (int i = tid; i < BT * EE / 2; i += 256) ((unsigned int*)hbf)[i] = 0u;
    if (tid < BT) logp[tid] = 0.0f;

    // ---- x0 = pos_encode(pos[:,0]) (first LSTM call, zero state) ----
    {
        const int m = tid >> 3, j = tid & 7;
        const float p = (float)pos_list[(r0 + m) * DD + 0];
        #pragma unroll
        for (int i = 0; i < 32; ++i) {
            int e = j * 32 + i;
            int k = (e < 128) ? e : (e - 128);
            float ang = p * __expf(-(float)k * (9.210340371976184f / 128.0f));
            xbf[m][e] = f2bf((e < 128) ? __sinf(ang) : __cosf(ang));
        }
    }
    __syncthreads();

    // ---- LSTM cell: gates = x@W_ih^T + h@W_hh^T + b ----
    // wave w owns gate columns [32w,32w+32): pass tp handles cols 32w+16tp..+15,
    // mm is inner so each B-fragment feeds 2 WMMAs (M-reuse). h_new goes to the
    // staging half of buf, so no read/write hazard on hbf inside the passes.
    auto lstm_step = [&]() {
        #pragma unroll
        for (int tp = 0; tp < 2; ++tp) {
            v8f z = {0.f,0.f,0.f,0.f,0.f,0.f,0.f,0.f};
            v8f acc[4][2];
            #pragma unroll
            for (int q = 0; q < 4; ++q) { acc[q][0] = z; acc[q][1] = z; }
            for (int ks = 0; ks < 8; ++ks) {
                const int k0 = ks * 32;
                Frag ax[2], ah[2];
                #pragma unroll
                for (int mm = 0; mm < 2; ++mm) {
                    const int r = mm * 16 + am;
                    ax[mm].q[0] = *(const u32x4*)&xbf[r][k0 + half * 8];
                    ax[mm].q[1] = *(const u32x4*)&xbf[r][k0 + 16 + half * 8];
                    ah[mm].q[0] = *(const u32x4*)&hbf[r][k0 + half * 8];
                    ah[mm].q[1] = *(const u32x4*)&hbf[r][k0 + 16 + half * 8];
                }
                #pragma unroll
                for (int q = 0; q < 4; ++q) {
                    const int n = q * 256 + wv * 32 + tp * 16 + am;  // gate output idx
                    const unsigned short* bi = wih_bf + (size_t)n * EE + k0 + half * 8;
                    Frag b; b.q[0] = *(const u32x4*)bi; b.q[1] = *(const u32x4*)(bi + 16);
                    #pragma unroll
                    for (int mm = 0; mm < 2; ++mm)
                        acc[q][mm] = wmma_bf16(ax[mm].v, b.v, acc[q][mm]);
                    const unsigned short* bh = whh_bf + (size_t)n * EE + k0 + half * 8;
                    Frag b2f; b2f.q[0] = *(const u32x4*)bh; b2f.q[1] = *(const u32x4*)(bh + 16);
                    #pragma unroll
                    for (int mm = 0; mm < 2; ++mm)
                        acc[q][mm] = wmma_bf16(ah[mm].v, b2f.v, acc[q][mm]);
                }
            }
            #pragma unroll
            for (int mm = 0; mm < 2; ++mm)
            #pragma unroll
            for (int v = 0; v < 8; ++v) {
                const int m  = mm * 16 + v + 8 * half;
                const int cg = wv * 32 + tp * 16 + am;     // col 0..255 within each gate
                float gi = acc[0][mm][v] + bsum[       cg];
                float gf = acc[1][mm][v] + bsum[256 + cg];
                float gg = acc[2][mm][v] + bsum[512 + cg];
                float go = acc[3][mm][v] + bsum[768 + cg];
                float iv = sigf(gi), fv = sigf(gf), gv = tanhfast(gg), ov = sigf(go);
                float cn = fv * c_reg[tp][mm][v] + iv * gv;
                c_reg[tp][mm][v] = cn;
                hnew[m][cg] = f2bf(ov * tanhfast(cn));
            }
        }
        __syncthreads();   // all h_new staged
        for (int i = tid; i < BT * EE / 2; i += 256)
            ((unsigned int*)hbf)[i] = ((const unsigned int*)hnew)[i];
        __syncthreads();
    };

    lstm_step();   // initial lstm(pe[:,0], h=c=0)

    for (int step = 0; step < DD; ++step) {
        if (tid < BT) s_lds[tid] = samples[(r0 + tid) * DD + step];

        // ---- GEMM1: hid = relu(h @ W1^T + b1)  [32 x 512], 4 N-tiles x 2 M-tiles/wave ----
        {
            v8f z = {0.f,0.f,0.f,0.f,0.f,0.f,0.f,0.f};
            v8f acc[4][2];
            #pragma unroll
            for (int j = 0; j < 4; ++j) { acc[j][0] = z; acc[j][1] = z; }
            for (int ks = 0; ks < 8; ++ks) {
                const int k0 = ks * 32;
                Frag a[2];
                #pragma unroll
                for (int mm = 0; mm < 2; ++mm) {
                    const int r = mm * 16 + am;
                    a[mm].q[0] = *(const u32x4*)&hbf[r][k0 + half * 8];
                    a[mm].q[1] = *(const u32x4*)&hbf[r][k0 + 16 + half * 8];
                }
                #pragma unroll
                for (int j = 0; j < 4; ++j) {
                    const int n = wv * 64 + j * 16 + am;
                    const unsigned short* bp = w1_bf + (size_t)n * EE + k0 + half * 8;
                    Frag b; b.q[0] = *(const u32x4*)bp; b.q[1] = *(const u32x4*)(bp + 16);
                    #pragma unroll
                    for (int mm = 0; mm < 2; ++mm)
                        acc[j][mm] = wmma_bf16(a[mm].v, b.v, acc[j][mm]);
                }
            }
            #pragma unroll
            for (int j = 0; j < 4; ++j)
            #pragma unroll
            for (int mm = 0; mm < 2; ++mm)
            #pragma unroll
            for (int v = 0; v < 8; ++v) {
                const int m   = mm * 16 + v + 8 * half;
                const int col = wv * 64 + j * 16 + am;
                hidbf[m][col] = f2bf(fmaxf(acc[j][mm][v] + b1s[col], 0.0f));
            }
        }
        __syncthreads();

        // ---- GEMM2: logits = hid @ W2^T + b2  [32 x 256], 2 N-tiles x 2 M-tiles/wave ----
        {
            v8f z = {0.f,0.f,0.f,0.f,0.f,0.f,0.f,0.f};
            v8f acc[2][2];
            #pragma unroll
            for (int t = 0; t < 2; ++t) { acc[t][0] = z; acc[t][1] = z; }
            for (int ks = 0; ks < 16; ++ks) {
                const int k0 = ks * 32;
                Frag a[2];
                #pragma unroll
                for (int mm = 0; mm < 2; ++mm) {
                    const int r = mm * 16 + am;
                    a[mm].q[0] = *(const u32x4*)&hidbf[r][k0 + half * 8];
                    a[mm].q[1] = *(const u32x4*)&hidbf[r][k0 + 16 + half * 8];
                }
                #pragma unroll
                for (int t = 0; t < 2; ++t) {
                    const int n = wv * 32 + t * 16 + am;
                    const unsigned short* bp = w2_bf + (size_t)n * H2 + k0 + half * 8;
                    Frag b; b.q[0] = *(const u32x4*)bp; b.q[1] = *(const u32x4*)(bp + 16);
                    #pragma unroll
                    for (int mm = 0; mm < 2; ++mm)
                        acc[t][mm] = wmma_bf16(a[mm].v, b.v, acc[t][mm]);
                }
            }
            __syncthreads();   // hidbf reads done before logits overwrite same LDS
            #pragma unroll
            for (int t = 0; t < 2; ++t)
            #pragma unroll
            for (int mm = 0; mm < 2; ++mm)
            #pragma unroll
            for (int v = 0; v < 8; ++v) {
                const int m   = mm * 16 + v + 8 * half;
                const int col = wv * 32 + t * 16 + am;
                logits[m][col] = acc[t][mm][v] + b2s[col];
            }
        }
        __syncthreads();

        // ---- log_softmax + gather: logp[r] += logits[r][s] - max - log(sum exp) ----
        {
            const int m = tid >> 3, j = tid & 7;
            float mx = -1e30f;
            #pragma unroll
            for (int i = 0; i < 32; ++i) mx = fmaxf(mx, logits[m][j * 32 + i]);
            red[m][j] = mx;
        }
        __syncthreads();
        if (tid < BT) {
            float mx = -1e30f;
            #pragma unroll
            for (int j = 0; j < 8; ++j) mx = fmaxf(mx, red[tid][j]);
            rowmax[tid] = mx;
        }
        __syncthreads();
        {
            const int m = tid >> 3, j = tid & 7;
            float s = 0.0f, mx = rowmax[m];
            #pragma unroll
            for (int i = 0; i < 32; ++i) s += __expf(logits[m][j * 32 + i] - mx);
            red[m][j] = s;
        }
        __syncthreads();
        if (tid < BT) {
            float s = 0.0f;
            #pragma unroll
            for (int j = 0; j < 8; ++j) s += red[tid][j];
            logp[tid] += logits[tid][s_lds[tid]] - rowmax[tid] - __logf(s);
        }
        __syncthreads();   // logits fully consumed; buf lo can be reused for x

        // ---- x = token_embed[s] + pos_encode(step) (into buf lo 16KB) ----
        {
            const int m = tid >> 3, j = tid & 7;
            const float p = (float)pos_list[(r0 + m) * DD + step];
            const int s = s_lds[m];
            #pragma unroll
            for (int i = 0; i < 32; ++i) {
                int e = j * 32 + i;
                int k = (e < 128) ? e : (e - 128);
                float ang = p * __expf(-(float)k * (9.210340371976184f / 128.0f));
                float pe  = (e < 128) ? __sinf(ang) : __cosf(ang);
                xbf[m][e] = f2bf(bf2f(tok_bf[s * EE + e]) + pe);
            }
        }
        __syncthreads();

        lstm_step();
    }

    if (tid < BT) out[r0 + tid] = logp[tid];
}

extern "C" void kernel_launch(void* const* d_in, const int* in_sizes, int n_in,
                              void* d_out, int out_size, void* d_ws, size_t ws_size,
                              hipStream_t stream) {
    const float* tok  = (const float*)d_in[0];
    const float* wih  = (const float*)d_in[1];
    const float* bih  = (const float*)d_in[2];
    const float* whh  = (const float*)d_in[3];
    const float* bhh  = (const float*)d_in[4];
    const float* w1   = (const float*)d_in[5];
    const float* b1   = (const float*)d_in[6];
    const float* w2   = (const float*)d_in[7];
    const float* b2   = (const float*)d_in[8];
    const int*   pos  = (const int*)d_in[9];
    const int*   smp  = (const int*)d_in[10];
    float*       out  = (float*)d_out;

    unsigned short* ws = (unsigned short*)d_ws;

    cvt_weights<<<(WS_TOT + 255) / 256, 256, 0, stream>>>(wih, whh, w1, w2, tok, ws);

    rnn_persistent<<<BB / BT, 256, 0, stream>>>(
        ws + OFF_WIH, ws + OFF_WHH, ws + OFF_W1, ws + OFF_W2, ws + OFF_TOK,
        bih, bhh, b1, b2, pos, smp, out);
}
